// meta_Net_DNN_8555574853902
// MI455X (gfx1250) — matrix operand order repaired and checked by reading, not compiled, not run on hardware
//
#include <hip/hip_runtime.h>
#include <hip/hip_bf16.h>
#include <math.h>

// ---------------------------------------------------------------------------
// Feature detection: Tensor Data Mover builtin (arity differs per toolchain)
// ---------------------------------------------------------------------------
#ifdef __has_builtin
#if __has_builtin(__builtin_amdgcn_tensor_load_to_lds)
#define HAVE_TDM 1
#endif
#endif
#ifndef HAVE_TDM
#define HAVE_TDM 0
#endif

#if defined(__has_include)
#if __has_include(<hip/amd_detail/amd_gfx1250_TDM.h>)
#define TDM_6ARG 1   // therock headers -> 6-arg builtin
#endif
#endif
#ifndef TDM_6ARG
#define TDM_6ARG 0   // ROCm 7.2 -> 5-arg builtin
#endif

// ---------------------------------------------------------------------------
// Types
// ---------------------------------------------------------------------------
typedef __attribute__((ext_vector_type(16))) _Float16 v16h;
typedef __attribute__((ext_vector_type(8)))  _Float16 v8h;
typedef __attribute__((ext_vector_type(4)))  _Float16 v4h;
typedef __attribute__((ext_vector_type(8)))  float    v8f;
typedef __attribute__((ext_vector_type(4)))  unsigned int u32x4;
typedef __attribute__((ext_vector_type(8)))  int      i32x8;
typedef __attribute__((ext_vector_type(4)))  int      i32x4;

#define ACT_NONE 0
#define ACT_RELU 1
#define ACT_TANH 2

// ---------------------------------------------------------------------------
// f32 -> f16 pack
// ---------------------------------------------------------------------------
__global__ __launch_bounds__(256) void cvt_f32_f16(const float* __restrict__ src,
                                                   _Float16* __restrict__ dst,
                                                   int n4) {
    int i = blockIdx.x * blockDim.x + threadIdx.x;
    int stride = gridDim.x * blockDim.x;
    for (; i < n4; i += stride) {
        float4 v = ((const float4*)src)[i];
        v4h o;
        o.x = (_Float16)v.x; o.y = (_Float16)v.y;
        o.z = (_Float16)v.z; o.w = (_Float16)v.w;
        ((v4h*)dst)[i] = o;
    }
}

// ---------------------------------------------------------------------------
// TDM helper: DMA a rows x 32 tile of f16 (row stride = K elements) from
// global memory into LDS at byte offset lds_off.  D# per CDNA5 ISA ch.8:
//   group0: count=1 | lds_addr[63:32] | global_addr[120:64] | type=2[127:126]
//   group1: data_size=2B | tensor_dim0=K | tensor_dim1=rows_total |
//           tile_dim0=32 | tile_dim1=rows | tensor_dim0_stride=K
// ---------------------------------------------------------------------------
#if HAVE_TDM
__device__ __forceinline__ void tdm_load_tile(unsigned lds_off,
                                              const _Float16* gptr,
                                              int tile_rows, int K,
                                              int tensor_rows) {
    unsigned long long ga = (unsigned long long)(size_t)gptr;
    u32x4 g0;
    g0[0] = 1u;                                   // count=1, user mode
    g0[1] = lds_off;                              // lds_addr
    g0[2] = (unsigned)(ga & 0xffffffffull);       // global_addr[31:0]
    g0[3] = (unsigned)((ga >> 32) & 0x01ffffffull)// global_addr[56:32]
          | (2u << 30);                           // type = 2 ("image")
    i32x8 g1;
    g1[0] = 0x00010000;                           // wg_mask=0, data_size=2B
    g1[1] = (int)(((unsigned)K & 0xffffu) << 16);               // dim0 lo16
    g1[2] = (int)((((unsigned)K >> 16) & 0xffffu)
          | (((unsigned)tensor_rows & 0xffffu) << 16));         // dim0 hi|dim1 lo
    g1[3] = (int)((((unsigned)tensor_rows >> 16) & 0xffffu)
          | (32u << 16));                                       // dim1 hi|tile_dim0
    g1[4] = (int)((unsigned)tile_rows & 0xffffu);               // tile_dim1
    g1[5] = K;                                                  // dim0_stride lo32
    g1[6] = 0;                                                  // dim0_stride hi16
    g1[7] = 0;
    i32x4 z4 = {0, 0, 0, 0};
#if TDM_6ARG
    i32x8 z8 = {0, 0, 0, 0, 0, 0, 0, 0};
    __builtin_amdgcn_tensor_load_to_lds(g0, g1, z4, z4, z8, 0);
#else
    __builtin_amdgcn_tensor_load_to_lds(g0, g1, z4, z4, 0);
#endif
}

__device__ __forceinline__ unsigned lds_byte_off(const void* p) {
    // generic pointers to LDS carry the LDS byte offset in their low 32 bits
    return (unsigned)(unsigned long long)(size_t)p;
}
#endif  // HAVE_TDM

// ---------------------------------------------------------------------------
// WMMA GEMM:  out[M,N] = act( A[M,K] @ W[N,K]^T + bias[N] )
// Block: 256 threads (8 wave32), block tile 128(M) x 64(N), K-step 32.
// Double-buffered LDS; tiles DMA'd by the Tensor Data Mover (wave 0 issues,
// TENSORcnt-pipelined one step ahead).  Each wave computes a 32x32 patch via
// 4 x v_wmma_f32_16x16x32_f16 per K-step.
// Requires: M % 128 == 0, N % 64 == 0, K % 32 == 0.
// ---------------------------------------------------------------------------
template<int ACT, bool HALF_OUT>
__global__ __launch_bounds__(256) void gemm_bias_act(
        const _Float16* __restrict__ A, const _Float16* __restrict__ W,
        const float* __restrict__ bias, void* __restrict__ out,
        int M, int N, int K)
{
    constexpr int BM = 128, BN = 64, KT = 32;
    __shared__ __align__(16) _Float16 Alds[2][BM * KT];
    __shared__ __align__(16) _Float16 Blds[2][BN * KT];

    const int tid  = threadIdx.x;
    const int lane = tid & 31;
    const int wave = tid >> 5;
    const int wm   = (wave & 3) << 5;   // 0,32,64,96  (M offset in block tile)
    const int wn   = (wave >> 2) << 5;  // 0,32        (N offset in block tile)
    const int m_blk = blockIdx.y * BM;
    const int n_blk = blockIdx.x * BN;

    const int lr   = lane & 15;
    const int hsel = lane >> 4;

    const int nk = K / KT;
    v8f acc[2][2] = {};

#if HAVE_TDM
    const unsigned ldsA[2] = { lds_byte_off(&Alds[0][0]), lds_byte_off(&Alds[1][0]) };
    const unsigned ldsB[2] = { lds_byte_off(&Blds[0][0]), lds_byte_off(&Blds[1][0]) };
    if (wave == 0) {   // prologue: DMA tiles for k-step 0 into buffer 0
        tdm_load_tile(ldsA[0], A + (size_t)m_blk * K, BM, K, M);
        tdm_load_tile(ldsB[0], W + (size_t)n_blk * K, BN, K, N);
    }
#else
    const int ar0 = tid >> 2,         ao0 = (tid & 3) * 8;
    const int ar1 = (tid + 256) >> 2, ao1 = ((tid + 256) & 3) * 8;
    const int br  = tid >> 2,         bo  = (tid & 3) * 8;
#endif

    for (int k = 0; k < nk; ++k) {
        const int buf = k & 1;
#if HAVE_TDM
        if (wave == 0) {
            if (k + 1 < nk) {   // DMA next step's tiles into the other buffer
                const int kk = (k + 1) * KT;
                tdm_load_tile(ldsA[buf ^ 1], A + (size_t)m_blk * K + kk, BM, K, M);
                tdm_load_tile(ldsB[buf ^ 1], W + (size_t)n_blk * K + kk, BN, K, N);
                __builtin_amdgcn_s_wait_tensorcnt(2);  // current buffer done
            } else {
                __builtin_amdgcn_s_wait_tensorcnt(0);
            }
        }
        __syncthreads();
#else
        {
            const int kk = k * KT;
            *(v8h*)&Alds[buf][ar0 * KT + ao0] =
                *(const v8h*)&A[(size_t)(m_blk + ar0) * K + kk + ao0];
            *(v8h*)&Alds[buf][ar1 * KT + ao1] =
                *(const v8h*)&A[(size_t)(m_blk + ar1) * K + kk + ao1];
            *(v8h*)&Blds[buf][br * KT + bo] =
                *(const v8h*)&W[(size_t)(n_blk + br) * K + kk + bo];
            if (kk + KT < K) {
                __builtin_prefetch(&A[(size_t)(m_blk + ar0) * K + kk + KT + ao0], 0, 1);
                __builtin_prefetch(&W[(size_t)(n_blk + br) * K + kk + KT + bo], 0, 1);
            }
        }
        __syncthreads();
#endif

        // ---- fragments from LDS per CDNA5 WMMA 16-bit layouts ----
        v16h afrag[2], bfrag[2];
#pragma unroll
        for (int i = 0; i < 2; ++i) {
            const _Float16* pa = &Alds[buf][(wm + i * 16 + lr) * KT + hsel * 8];
            v8h alo = *(const v8h*)pa;
            v8h ahi = *(const v8h*)(pa + 16);
            afrag[i] = __builtin_shufflevector(alo, ahi,
                0,1,2,3,4,5,6,7,8,9,10,11,12,13,14,15);
            const _Float16* pb = &Blds[buf][(wn + i * 16 + lr) * KT + hsel * 8];
            v8h blo = *(const v8h*)pb;
            v8h bhi = *(const v8h*)(pb + 16);
            bfrag[i] = __builtin_shufflevector(blo, bhi,
                0,1,2,3,4,5,6,7,8,9,10,11,12,13,14,15);
        }

#pragma unroll
        for (int i = 0; i < 2; ++i)
#pragma unroll
            for (int j = 0; j < 2; ++j)
                acc[i][j] = __builtin_amdgcn_wmma_f32_16x16x32_f16(
                    false, afrag[i], false, bfrag[j],
                    (short)0, acc[i][j], false, false);
        __syncthreads();  // all reads done before this buffer is overwritten
    }

    // ---- store: C/D 16x16 f32 layout: VGPR r, lanes 0-15 -> M=r, 16-31 -> M=8+r
#pragma unroll
    for (int i = 0; i < 2; ++i) {
#pragma unroll
        for (int j = 0; j < 2; ++j) {
            const int col  = n_blk + wn + j * 16 + lr;
            const float bv = bias[col];
            const int mrow = m_blk + wm + i * 16 + (hsel << 3);
#pragma unroll
            for (int r = 0; r < 8; ++r) {
                float v = acc[i][j][r] + bv;
                if (ACT == ACT_RELU) v = v > 0.f ? v : 0.f;
                if (ACT == ACT_TANH) v = tanhf(v);
                const size_t idx = (size_t)(mrow + r) * N + col;
                if (HALF_OUT) ((_Float16*)out)[idx] = (_Float16)v;
                else          ((float*)out)[idx]    = v;
            }
        }
    }
}

// ---------------------------------------------------------------------------
// Channel kernel: one block (128 threads) per batch row, one complex symbol
// per thread.  Power norm -> fading channel + AWGN -> LS estimate -> ZF.
// ---------------------------------------------------------------------------
__global__ __launch_bounds__(128) void channel_kernel(
        const float* __restrict__ s, const float* __restrict__ h,
        const float* __restrict__ noise,
        float* __restrict__ xr_f, _Float16* __restrict__ xr_h)
{
    constexpr int CH = 256;
    const int b = blockIdx.x;
    const int t = threadIdx.x;

    __shared__ float r0[128], r1[128], r2[128];

    const float* srow = s + (size_t)b * CH;
    float re = srow[2 * t], im = srow[2 * t + 1];

    r0[t] = re * re + im * im;
    __syncthreads();
    for (int o = 64; o > 0; o >>= 1) {
        if (t < o) r0[t] += r0[t + o];
        __syncthreads();
    }
    const float nrm2 = r0[0];
    __syncthreads();

    const float scale = sqrtf((float)CH * 0.5f) * rsqrtf(nrm2);
    const float txr = scale * re, txi = scale * im;

    const float hr = h[0], hi = h[1];
    const float* nrow = noise + (size_t)b * CH;
    const float yr = hr * txr - hi * txi + nrow[2 * t];
    const float yi = hr * txi + hi * txr + nrow[2 * t + 1];

    r0[t] = yr * txr + yi * txi;
    r1[t] = yi * txr - yr * txi;
    r2[t] = txr * txr + txi * txi;
    __syncthreads();
    for (int o = 64; o > 0; o >>= 1) {
        if (t < o) { r0[t] += r0[t + o]; r1[t] += r1[t + o]; r2[t] += r2[t + o]; }
        __syncthreads();
    }
    const float den = r2[0];
    const float her = r0[0] / den, hei = r1[0] / den;
    const float hm2 = her * her + hei * hei + 1e-12f;

    const float xer = (yr * her + yi * hei) / hm2;
    const float xei = (yi * her - yr * hei) / hm2;

    const size_t base = (size_t)b * CH;
    xr_f[base + 2 * t]     = xer;
    xr_f[base + 2 * t + 1] = xei;
    xr_h[base + 2 * t]     = (_Float16)xer;
    xr_h[base + 2 * t + 1] = (_Float16)xei;
}

// ---------------------------------------------------------------------------
// RTN tail: per row, h_inv = g2 @ w_rtn3^T + b_rtn3 (6 dots of length 1024),
// then 3-tap causal complex FIR over xr (128 complex samples).
// ---------------------------------------------------------------------------
__global__ __launch_bounds__(256) void rtn_tail_kernel(
        const _Float16* __restrict__ g2, const float* __restrict__ w_rtn3,
        const float* __restrict__ b_rtn3, const float* __restrict__ xr_f,
        _Float16* __restrict__ xr2_h)
{
    constexpr int HR = 1024, CH = 256, L = 3;
    const int b = blockIdx.x;
    const int t = threadIdx.x;

    __shared__ float red[6][256];
    __shared__ float hinv[6];

    const _Float16* grow = g2 + (size_t)b * HR;
    float p[6] = {0.f, 0.f, 0.f, 0.f, 0.f, 0.f};
    for (int k = t; k < HR; k += 256) {
        const float gv = (float)grow[k];
#pragma unroll
        for (int j = 0; j < 6; ++j) p[j] += gv * w_rtn3[j * HR + k];
    }
#pragma unroll
    for (int j = 0; j < 6; ++j) red[j][t] = p[j];
    __syncthreads();
    for (int o = 128; o > 0; o >>= 1) {
        if (t < o) {
#pragma unroll
            for (int j = 0; j < 6; ++j) red[j][t] += red[j][t + o];
        }
        __syncthreads();
    }
    if (t < 6) hinv[t] = red[t][0] + b_rtn3[t];
    __syncthreads();

    if (t < CH / 2) {
        const float* xrow = xr_f + (size_t)b * CH;
        float outr = 0.f, outi = 0.f;
#pragma unroll
        for (int l = 0; l < L; ++l) {
            const int sidx = t - l;
            if (sidx >= 0) {
                const float xr = xrow[2 * sidx], xi = xrow[2 * sidx + 1];
                const float hr = hinv[2 * l],    hi = hinv[2 * l + 1];
                outr += hr * xr - hi * xi;
                outi += hr * xi + hi * xr;
            }
        }
        xr2_h[(size_t)b * CH + 2 * t]     = (_Float16)outr;
        xr2_h[(size_t)b * CH + 2 * t + 1] = (_Float16)outi;
    }
}

// ---------------------------------------------------------------------------
// Host launcher
// ---------------------------------------------------------------------------
extern "C" void kernel_launch(void* const* d_in, const int* in_sizes, int n_in,
                              void* d_out, int out_size, void* d_ws, size_t ws_size,
                              hipStream_t stream) {
    constexpr int B = 8192, D_IN = 512, H1 = 4096, CH = 256, HR = 1024,
                  H2 = 4096, D_OUT = 512;

    const float* x      = (const float*)d_in[0];
    const float* w1     = (const float*)d_in[1];
    const float* b1     = (const float*)d_in[2];
    const float* w2     = (const float*)d_in[3];
    const float* b2     = (const float*)d_in[4];
    const float* w_rtn1 = (const float*)d_in[5];
    const float* b_rtn1 = (const float*)d_in[6];
    const float* w_rtn2 = (const float*)d_in[7];
    const float* b_rtn2 = (const float*)d_in[8];
    const float* w_rtn3 = (const float*)d_in[9];
    const float* b_rtn3 = (const float*)d_in[10];
    const float* w3     = (const float*)d_in[11];
    const float* b3     = (const float*)d_in[12];
    const float* w4     = (const float*)d_in[13];
    const float* b4     = (const float*)d_in[14];
    const float* h      = (const float*)d_in[15];
    const float* noise  = (const float*)d_in[16];

    char* ws = (char*)d_ws;
    size_t off = 0;
    auto carve = [&](size_t bytes) -> char* {
        char* p = ws + off;
        off = (off + bytes + 255) & ~(size_t)255;
        return p;
    };
    _Float16* xh    = (_Float16*)carve((size_t)B * D_IN * 2);
    _Float16* w1h   = (_Float16*)carve((size_t)H1 * D_IN * 2);
    _Float16* w2h   = (_Float16*)carve((size_t)CH * H1 * 2);
    _Float16* wr1h  = (_Float16*)carve((size_t)HR * CH * 2);
    _Float16* wr2h  = (_Float16*)carve((size_t)HR * HR * 2);
    _Float16* w3h   = (_Float16*)carve((size_t)H2 * CH * 2);
    _Float16* w4h   = (_Float16*)carve((size_t)D_OUT * H2 * 2);
    float*    s_f   = (float*)carve((size_t)B * CH * 4);
    float*    xr_f  = (float*)carve((size_t)B * CH * 4);
    _Float16* xr_h  = (_Float16*)carve((size_t)B * CH * 2);
    _Float16* xr2_h = (_Float16*)carve((size_t)B * CH * 2);
    _Float16* big   = (_Float16*)carve((size_t)B * H1 * 2);
    _Float16* a_h   = big;
    _Float16* g1_h  = big;                       // a dead once s computed
    _Float16* g2_h  = big + (size_t)B * HR;
    _Float16* h3_h  = big;                       // g1/g2 dead after rtn_tail
    (void)ws_size; (void)in_sizes; (void)n_in; (void)out_size;

    auto cvt = [&](const float* src, _Float16* dst, size_t n) {
        int n4 = (int)(n / 4);
        int blocks = (n4 + 255) / 256;
        if (blocks > 4096) blocks = 4096;
        cvt_f32_f16<<<blocks, 256, 0, stream>>>(src, dst, n4);
    };
    cvt(x,      xh,   (size_t)B * D_IN);
    cvt(w1,     w1h,  (size_t)H1 * D_IN);
    cvt(w2,     w2h,  (size_t)CH * H1);
    cvt(w_rtn1, wr1h, (size_t)HR * CH);
    cvt(w_rtn2, wr2h, (size_t)HR * HR);
    cvt(w3,     w3h,  (size_t)H2 * CH);
    cvt(w4,     w4h,  (size_t)D_OUT * H2);

    auto grid = [](int M, int N) { return dim3((unsigned)(N / 64), (unsigned)(M / 128)); };

    gemm_bias_act<ACT_RELU, true><<<grid(B, H1), 256, 0, stream>>>(
        xh, w1h, b1, (void*)a_h, B, H1, D_IN);
    gemm_bias_act<ACT_NONE, false><<<grid(B, CH), 256, 0, stream>>>(
        a_h, w2h, b2, (void*)s_f, B, CH, H1);
    channel_kernel<<<B, 128, 0, stream>>>(s_f, h, noise, xr_f, xr_h);
    gemm_bias_act<ACT_TANH, true><<<grid(B, HR), 256, 0, stream>>>(
        xr_h, wr1h, b_rtn1, (void*)g1_h, B, HR, CH);
    gemm_bias_act<ACT_TANH, true><<<grid(B, HR), 256, 0, stream>>>(
        g1_h, wr2h, b_rtn2, (void*)g2_h, B, HR, HR);
    rtn_tail_kernel<<<B, 256, 0, stream>>>(g2_h, w_rtn3, b_rtn3, xr_f, xr2_h);
    gemm_bias_act<ACT_RELU, true><<<grid(B, H2), 256, 0, stream>>>(
        xr2_h, w3h, b3, (void*)h3_h, B, H2, CH);
    gemm_bias_act<ACT_NONE, false><<<grid(B, D_OUT), 256, 0, stream>>>(
        h3_h, w4h, b4, d_out, B, D_OUT, H2);
}
